// SpectralLinear_24086176596622
// MI455X (gfx1250) — compile-verified, matrix operand order checked by reading, not run on hardware
//
#include <hip/hip_runtime.h>
#include <hip/hip_bf16.h>

// SpectralLinear on MI455X (gfx1250, wave32).
//
// out = x @ (H_U0..H_U63) @ Sigma @ (H_V63..H_V0) + bias
//   -> collapse the whole operator into W (512x512), then one WMMA GEMM.
//
// Kernel 1 (house_build): one wave per row of W; the Householder update only
//   touches that row, so all 128 reflections run with zero inter-wave sync.
//   W is stored directly as a bf16 hi/lo split, transposed and K-contiguous
//   per output column (= WMMA B-operand memory order): zero B-side conversion
//   in the GEMM.
//
// Kernel 2 (wmma GEMM): out[8192x512] = x @ W + bias via
//   v_wmma_f32_16x16x32_bf16, 3-term bf16 split (Xh*Wh + Xh*Wl + Xl*Wh).
//   The A (x) hi/lo split is staged cooperatively through LDS once per block
//   (256 threads convert 512 floats, double-buffered, one workgroup barrier
//   per k-chunk) instead of redundantly in every wave: the loop body is now
//   loads + 12 WMMA with ~6 conversion VALU per thread, i.e. memory-bound
//   (~34 MB HBM traffic ~ 1.5 us at 23.3 TB/s).

typedef __attribute__((ext_vector_type(16))) __bf16 v16bf;
typedef __attribute__((ext_vector_type(8)))  __bf16 v8bf;   // one b128
typedef __attribute__((ext_vector_type(2)))  __bf16 v2bf;   // one b32
typedef __attribute__((ext_vector_type(8)))  float  v8f;
typedef __attribute__((ext_vector_type(2)))  float  vf2;

#define INSIZE   512
#define OUTSIZE  512
#define N_REFL   64
#define BATCH    8192

// ---------------------------------------------------------------------------
// Kernel 1: build W, store as bf16 hi/lo, transposed (column-major over k).
//   Wth[n*512 + k] = bf16_hi(W[k][n]);  Wtl likewise for the residual.
// ---------------------------------------------------------------------------
__global__ __launch_bounds__(256) void
spectral_house_build(const float* __restrict__ U,
                     const float* __restrict__ V,
                     const float* __restrict__ p,
                     __bf16* __restrict__ Wth,
                     __bf16* __restrict__ Wtl) {
  const int gtid = blockIdx.x * blockDim.x + threadIdx.x;
  const int row  = gtid >> 5;         // 0..511  (row j of W)
  const int lane = threadIdx.x & 31;

  // Row j of evolving matrix; element k = lane + 32*t lives in m[t].
  float m[16];
#pragma unroll
  for (int t = 0; t < 16; ++t)
    m[t] = ((lane + 32 * t) == row) ? 1.0f : 0.0f;

  // ---- U chain: i = 0 .. 63, reflector u = triu(U)[i] ----
  for (int i = 0; i < N_REFL; ++i) {
    const float* u = U + i * INSIZE;
    float uloc[16];
    float uu = 0.0f, mu = 0.0f;
#pragma unroll
    for (int t = 0; t < 16; ++t) {
      const int k = lane + 32 * t;
      const float uv = (k >= i) ? u[k] : 0.0f;   // upper-triangular mask
      uloc[t] = uv;
      uu += uv * uv;
      mu += m[t] * uv;
    }
#pragma unroll
    for (int off = 16; off >= 1; off >>= 1) {    // wave32 tree reduction
      uu += __shfl_xor(uu, off, 32);
      mu += __shfl_xor(mu, off, 32);
    }
    const float s = 2.0f * mu / uu;
#pragma unroll
    for (int t = 0; t < 16; ++t) m[t] -= s * uloc[t];
  }

  // ---- Sigma: sigmas = 0.9*(sigmoid(p)-0.5) + 0.55 (column scaling) ----
#pragma unroll
  for (int t = 0; t < 16; ++t) {
    const int k = lane + 32 * t;
    const float pv  = p[k];
    const float sig = 0.9f * (1.0f / (1.0f + __expf(-pv)) - 0.5f) + 0.55f;
    m[t] *= sig;
  }

  // ---- V chain: i = 63 down to 0, reflector v = triu(V)[i] ----
  for (int i = N_REFL - 1; i >= 0; --i) {
    const float* v = V + i * OUTSIZE;
    float uloc[16];
    float uu = 0.0f, mu = 0.0f;
#pragma unroll
    for (int t = 0; t < 16; ++t) {
      const int k = lane + 32 * t;
      const float uv = (k >= i) ? v[k] : 0.0f;
      uloc[t] = uv;
      uu += uv * uv;
      mu += m[t] * uv;
    }
#pragma unroll
    for (int off = 16; off >= 1; off >>= 1) {
      uu += __shfl_xor(uu, off, 32);
      mu += __shfl_xor(mu, off, 32);
    }
    const float s = 2.0f * mu / uu;
#pragma unroll
    for (int t = 0; t < 16; ++t) m[t] -= s * uloc[t];
  }

  // Store bf16 hi/lo, transposed: index [k * 512 + row].
#pragma unroll
  for (int t = 0; t < 16; ++t) {
    const int k = lane + 32 * t;
    const float f  = m[t];
    const __bf16 hi = (__bf16)f;
    const __bf16 lo = (__bf16)(f - (float)hi);
    Wth[k * OUTSIZE + row] = hi;
    Wtl[k * OUTSIZE + row] = lo;
  }
}

// ---------------------------------------------------------------------------
// Kernel 2: out[8192x512] = x @ W + bias  via v_wmma_f32_16x16x32_bf16.
// Block = 256 threads (8 waves) = one 16-row M tile; wave w covers columns
// [64w, 64w+64) as 4 accumulator tiles. K looped in chunks of 32.
// A split staged cooperatively through double-buffered LDS (one barrier per
// chunk); B operands are pre-split bf16 (two b128 loads each, no VALU).
// ---------------------------------------------------------------------------
__global__ __launch_bounds__(256) void
spectral_wmma_gemm(const float* __restrict__ X,
                   const __bf16* __restrict__ Wth,
                   const __bf16* __restrict__ Wtl,
                   const float* __restrict__ bias,
                   float* __restrict__ out) {
  __shared__ __bf16 sAh[2][16 * 32];   // A-hi tile, row-major (m, k) : 1 KB x2
  __shared__ __bf16 sAl[2][16 * 32];   // A-lo tile                  : 1 KB x2

  const int tid   = threadIdx.x;
  const int lane  = tid & 31;
  const int wave  = tid >> 5;           // 0..7
  const int mTile = blockIdx.x;         // 0..511
  const int n0    = wave * 64;
  const int mrow  = lane & 15;          // M (for A) / N (for B) index
  const int h     = lane >> 4;          // lane-half selects K sub-block

  // Stage assignment: thread converts elements idx, idx+1 of the 16x32 tile.
  const int idx = tid * 2;              // 0..510 (even)
  const int sr  = idx >> 5;             // source row 0..15
  const int sc  = idx & 31;             // source k offset (even)
  const float* xstage = X + (size_t)(mTile * 16 + sr) * INSIZE + sc;

  v8f acc[4];
  const v8f vz = {0.f, 0.f, 0.f, 0.f, 0.f, 0.f, 0.f, 0.f};
#pragma unroll
  for (int nt = 0; nt < 4; ++nt) acc[nt] = vz;

  for (int kc = 0; kc < INSIZE / 32; ++kc) {
    const int kbase = kc * 32;
    const int buf   = kc & 1;

    // ---- stage: cooperative A hi/lo split into LDS (2 floats / thread) ----
    {
      if (kc + 1 < INSIZE / 32)
        __builtin_prefetch(xstage + kbase + 32, 0, 3);  // global_prefetch_b8
      const vf2 fx = *(const vf2*)(xstage + kbase);
      const __bf16 h0 = (__bf16)fx[0];
      const __bf16 h1 = (__bf16)fx[1];
      v2bf hi, lo;
      hi[0] = h0;
      hi[1] = h1;
      lo[0] = (__bf16)(fx[0] - (float)h0);
      lo[1] = (__bf16)(fx[1] - (float)h1);
      *(v2bf*)&sAh[buf][idx] = hi;
      *(v2bf*)&sAl[buf][idx] = lo;
    }
    __syncthreads();   // also separates compute(kc-2) from this buffer reuse

    // ---- A operands from LDS: two ds_load_b128 per operand ----
    const __bf16* abh = &sAh[buf][mrow * 32 + 8 * h];
    const __bf16* abl = &sAl[buf][mrow * 32 + 8 * h];
    const v8bf ah0 = *(const v8bf*)abh;
    const v8bf ah1 = *(const v8bf*)(abh + 16);
    const v8bf al0 = *(const v8bf*)abl;
    const v8bf al1 = *(const v8bf*)(abl + 16);
    const v16bf Ah = __builtin_shufflevector(ah0, ah1,
        0, 1, 2, 3, 4, 5, 6, 7, 8, 9, 10, 11, 12, 13, 14, 15);
    const v16bf Al = __builtin_shufflevector(al0, al1,
        0, 1, 2, 3, 4, 5, 6, 7, 8, 9, 10, 11, 12, 13, 14, 15);

#pragma unroll
    for (int nt = 0; nt < 4; ++nt) {
      const int ncol = n0 + nt * 16 + mrow;
      const size_t base = (size_t)ncol * INSIZE + kbase + 8 * h;

      // B operands: pre-split bf16, two b128 loads each (K-runs 8h+0..7, 16+8h+0..7).
      const v8bf bh0 = *(const v8bf*)(Wth + base);
      const v8bf bh1 = *(const v8bf*)(Wth + base + 16);
      const v8bf bl0 = *(const v8bf*)(Wtl + base);
      const v8bf bl1 = *(const v8bf*)(Wtl + base + 16);
      const v16bf Bh = __builtin_shufflevector(bh0, bh1,
          0, 1, 2, 3, 4, 5, 6, 7, 8, 9, 10, 11, 12, 13, 14, 15);
      const v16bf Bl = __builtin_shufflevector(bl0, bl1,
          0, 1, 2, 3, 4, 5, 6, 7, 8, 9, 10, 11, 12, 13, 14, 15);

      // fp32-grade product via 3-term bf16 split
      acc[nt] = __builtin_amdgcn_wmma_f32_16x16x32_bf16(
          false, Ah, false, Bh, (short)0, acc[nt], false, false);
      acc[nt] = __builtin_amdgcn_wmma_f32_16x16x32_bf16(
          false, Ah, false, Bl, (short)0, acc[nt], false, false);
      acc[nt] = __builtin_amdgcn_wmma_f32_16x16x32_bf16(
          false, Al, false, Bh, (short)0, acc[nt], false, false);
    }
  }

  // Epilogue: C/D layout -> vgpr r, lane l : row = r + 8h, col = l & 15.
#pragma unroll
  for (int nt = 0; nt < 4; ++nt) {
    const int ncol = n0 + nt * 16 + mrow;
    const float bv = bias[ncol];
#pragma unroll
    for (int r = 0; r < 8; ++r) {
      const int row = mTile * 16 + r + 8 * h;
      out[(size_t)row * OUTSIZE + ncol] = acc[nt][r] + bv;
    }
  }
}

// ---------------------------------------------------------------------------
extern "C" void kernel_launch(void* const* d_in, const int* in_sizes, int n_in,
                              void* d_out, int out_size, void* d_ws, size_t ws_size,
                              hipStream_t stream) {
  const float* x    = (const float*)d_in[0];   // [8192, 512]
  const float* p    = (const float*)d_in[1];   // [512]
  const float* U    = (const float*)d_in[2];   // [512, 512]
  const float* V    = (const float*)d_in[3];   // [512, 512]
  const float* bias = (const float*)d_in[4];   // [1, 512]
  float* out = (float*)d_out;                  // [8192, 512]

  __bf16* Wth = (__bf16*)d_ws;                  // 512*512 bf16 = 512 KB
  __bf16* Wtl = Wth + (size_t)INSIZE * OUTSIZE; // next 512 KB

  // 512 waves (one per row of W): 64 blocks x 256 threads.
  spectral_house_build<<<64, 256, 0, stream>>>(U, V, p, Wth, Wtl);

  // 512 M-tiles, 8 waves each covering 64 columns.
  spectral_wmma_gemm<<<BATCH / 16, 256, 0, stream>>>(x, Wth, Wtl, bias, out);
}
